// KeyedConv2d_76794015252828
// MI455X (gfx1250) — matrix operand order repaired and checked by reading, not compiled, not run on hardware
//
#include <hip/hip_runtime.h>

typedef _Float16 half_t;
typedef __attribute__((ext_vector_type(16))) _Float16 v16h;
typedef __attribute__((ext_vector_type(8)))  float    v8f;

#define N_BATCH 64
#define C_IN    64
#define C_OUT   64
#define H_IMG   56
#define W_IMG   56
#define HW      (H_IMG * W_IMG)     // 3136
#define CHW     (C_IN * HW)         // 200704
#define D_IN    (CHW + 1)           // 200705
#define D_OUT   (C_OUT * HW + 1)    // 200705
#define KTOT    (C_IN * 9)          // 576
#define KCHUNKS (KTOT / 32)         // 18
#define CO_TILES (C_OUT / 16)       // 4
#define ROW_TILES (H_IMG / 4)       // 14 (4 rows = 224 positions per block)
#define WFRAG_HALVES (CO_TILES * KCHUNKS * 32 * 16)   // 36864 halves = 73728 B
#define WFRAG_V16    (WFRAG_HALVES / 16)              // 2304 v16h chunks
#define BLOCK_THREADS 224

// ---------------------------------------------------------------------------
// Kernel 1: z = Ainv_scale * x[:, Ainv_perm], written NHWC as f16
//   (coalesced stores; perm/scale/x gathers are L2-resident: 192MB L2)
// ---------------------------------------------------------------------------
__global__ void prep_z(const float* __restrict__ x,
                       const float* __restrict__ ainv_s,
                       const int*   __restrict__ ainv_p,
                       half_t* __restrict__ znhwc,
                       float*  __restrict__ zlast) {
    long idx = (long)blockIdx.x * blockDim.x + threadIdx.x;
    long body = (long)N_BATCH * CHW;
    if (idx < body) {
        int n  = (int)(idx / CHW);
        int r  = (int)(idx % CHW);
        int hw = r >> 6;            // / 64
        int ci = r & 63;            // % 64
        int d  = ci * HW + hw;      // CHW-order index into z
        znhwc[idx] = (half_t)(ainv_s[d] * x[(long)n * D_IN + ainv_p[d]]);
    } else if (idx < body + N_BATCH) {
        int n = (int)(idx - body);
        int d = D_IN - 1;
        zlast[n] = ainv_s[d] * x[(long)n * D_IN + ainv_p[d]];
    }
}

// ---------------------------------------------------------------------------
// Kernel 2: repack fp32 weights into WMMA A-fragment order with k = tap*64+ci
//   wfrag[((ct*KCHUNKS + kc)*32 + lane)*16 + j] = f16(w[co][ci][tap])
//   co = ct*16 + lane%16 ; k = kc*32 + 16*(lane/16) + j (contiguous order, same as B)
// ---------------------------------------------------------------------------
__global__ void prep_w(const float* __restrict__ w, half_t* __restrict__ wfrag) {
    int idx = blockIdx.x * blockDim.x + threadIdx.x;
    if (idx >= WFRAG_HALVES) return;
    int j    = idx & 15;
    int lane = (idx >> 4) & 31;
    int g    = idx >> 9;                 // ct*KCHUNKS + kc
    int kc   = g % KCHUNKS;
    int ct   = g / KCHUNKS;
    int koff = j;                        // lanes 0-15: K in {0..15}
    if (lane >= 16) koff += 16;          // lanes 16-31: K in {16..31}
    int co = ct * 16 + (lane & 15);
    int k  = kc * 32 + koff;
    int t  = k >> 6;                     // tap 0..8
    int ci = k & 63;
    wfrag[idx] = (half_t)w[(co * C_IN + ci) * 9 + t];  // w[co][ci][kh][kw]
}

// ---------------------------------------------------------------------------
// Kernel 3: implicit-GEMM conv via V_WMMA_F32_16X16X32_F16, NHWC in/out.
//   A fragments staged once per block in LDS (73,728 B of the 320KB WGP LDS),
//   then read as ds_load_b128 pairs. Each wave owns 32 spatial positions
//   (two 16-col tiles sharing A). Per k-chunk: 2 global B loads + 4 LDS A
//   loads + 8 accumulating WMMAs.
// ---------------------------------------------------------------------------
__global__ void __launch_bounds__(BLOCK_THREADS)
conv_wmma(const half_t* __restrict__ znhwc,
          const half_t* __restrict__ wfrag,
          float* __restrict__ y) {
    extern __shared__ char smem[];
    half_t* awf = (half_t*)smem;

    // Cooperative stage of weight fragments into LDS (global b128 -> ds b128)
    {
        const v16h* src = (const v16h*)wfrag;
        v16h* dst = (v16h*)awf;
        for (int i = threadIdx.x; i < WFRAG_V16; i += BLOCK_THREADS)
            dst[i] = src[i];
    }
    __syncthreads();

    int n    = blockIdx.x / ROW_TILES;
    int rt   = blockIdx.x % ROW_TILES;
    int lane = threadIdx.x & 31;
    int wave = threadIdx.x >> 5;                 // 0..6
    int col  = lane & 15;
    int base = rt * 224 + wave * 32;
    int pos0 = base + col;
    int pos1 = base + 16 + col;
    int h0 = pos0 / W_IMG, w0 = pos0 % W_IMG;
    int h1 = pos1 / W_IMG, w1 = pos1 % W_IMG;
    const half_t* zin = znhwc + (long)n * CHW;

    v8f acc0[CO_TILES] = {};
    v8f acc1[CO_TILES] = {};
    const v16h bzero = {};
    int khalf = (lane < 16) ? 0 : 16;            // ci offset within 32-chunk

    for (int t = 0; t < 9; ++t) {
        int dh = t / 3 - 1, dw = t % 3 - 1;
        int hh0 = h0 + dh, ww0 = w0 + dw;
        int hh1 = h1 + dh, ww1 = w1 + dw;
        bool in0 = (hh0 >= 0) & (hh0 < H_IMG) & (ww0 >= 0) & (ww0 < W_IMG);
        bool in1 = (hh1 >= 0) & (hh1 < H_IMG) & (ww1 >= 0) & (ww1 < W_IMG);
        const half_t* p0 = zin + (long)(hh0 * W_IMG + ww0) * C_IN + khalf;
        const half_t* p1 = zin + (long)(hh1 * W_IMG + ww1) * C_IN + khalf;
#pragma unroll
        for (int ch = 0; ch < 2; ++ch) {         // two 32-ci halves
            int kc = t * 2 + ch;
            v16h b0 = in0 ? *(const v16h*)(p0 + ch * 32) : bzero;
            v16h b1 = in1 ? *(const v16h*)(p1 + ch * 32) : bzero;
#pragma unroll
            for (int ct = 0; ct < CO_TILES; ++ct) {
                v16h a = *(const v16h*)(awf + ((ct * KCHUNKS + kc) * 32 + lane) * 16);
                acc0[ct] = __builtin_amdgcn_wmma_f32_16x16x32_f16(
                    false, a, false, b0, (short)0, acc0[ct], false, false);
                acc1[ct] = __builtin_amdgcn_wmma_f32_16x16x32_f16(
                    false, a, false, b1, (short)0, acc1[ct], false, false);
            }
        }
    }

    // C/D layout: VGPR r -> co = mbase + r (contiguous) => one v8f store each
    int mbase = (lane < 16) ? 0 : 8;
    float* y0 = y + ((long)n * HW + pos0) * C_OUT + mbase;
    float* y1 = y + ((long)n * HW + pos1) * C_OUT + mbase;
#pragma unroll
    for (int ct = 0; ct < CO_TILES; ++ct) {
        *(v8f*)(y0 + ct * 16) = acc0[ct];
        *(v8f*)(y1 + ct * 16) = acc1[ct];
    }
}

// ---------------------------------------------------------------------------
// Kernel 4: out = A_scale * (y_affine[:, A_perm]) with fused bias term
//   y is NHWC: element j (CHW order) -> y[(n*HW + j%HW)*64 + j/HW]
// ---------------------------------------------------------------------------
__global__ void out_perm(const float* __restrict__ y,
                         const float* __restrict__ zlast,
                         const float* __restrict__ b,
                         const float* __restrict__ a_s,
                         const int*   __restrict__ a_p,
                         float* __restrict__ out) {
    long idx = (long)blockIdx.x * blockDim.x + threadIdx.x;
    if (idx >= (long)N_BATCH * D_OUT) return;
    int n = (int)(idx / D_OUT);
    int q = (int)(idx % D_OUT);
    int j = a_p[q];
    float zl = zlast[n];
    float v;
    if (j == D_OUT - 1) {
        v = zl;
    } else {
        int co = j / HW;
        int hw = j - co * HW;
        v = y[((long)n * HW + hw) * C_OUT + co] + zl * b[co];
    }
    out[idx] = a_s[q] * v;
}

// ---------------------------------------------------------------------------
extern "C" void kernel_launch(void* const* d_in, const int* in_sizes, int n_in,
                              void* d_out, int out_size, void* d_ws, size_t ws_size,
                              hipStream_t stream) {
    const float* x      = (const float*)d_in[0];   // (N, D_IN)
    const float* w      = (const float*)d_in[1];   // (64,64,3,3)
    const float* b      = (const float*)d_in[2];   // (64,)
    const float* a_s    = (const float*)d_in[3];   // (D_OUT,)
    const float* ainv_s = (const float*)d_in[4];   // (D_IN,)
    const int*   a_p    = (const int*)d_in[5];     // (D_OUT,)
    const int*   ainv_p = (const int*)d_in[6];     // (D_IN,)
    float* out = (float*)d_out;

    // Workspace layout (256B-aligned offsets)
    char* ws = (char*)d_ws;
    half_t* znhwc = (half_t*)(ws);                              // 25,690,112 B
    half_t* wfrag = (half_t*)(ws + 25690112);                   //     73,728 B
    float*  zlast = (float*) (ws + 25690112 + 73728);           //        256 B
    float*  y     = (float*) (ws + 25690112 + 73728 + 256);     // 51,380,224 B

    {
        long total = (long)N_BATCH * CHW + N_BATCH;
        int blocks = (int)((total + 255) / 256);
        prep_z<<<blocks, 256, 0, stream>>>(x, ainv_s, ainv_p, znhwc, zlast);
    }
    {
        prep_w<<<(WFRAG_HALVES + 255) / 256, 256, 0, stream>>>(w, wfrag);
    }
    {
        size_t lds_bytes = (size_t)WFRAG_HALVES * sizeof(half_t);  // 73,728 B
        conv_wmma<<<N_BATCH * ROW_TILES, BLOCK_THREADS, lds_bytes, stream>>>(znhwc, wfrag, y);
    }
    {
        long total = (long)N_BATCH * D_OUT;
        int blocks = (int)((total + 255) / 256);
        out_perm<<<blocks, 256, 0, stream>>>(y, zlast, b, a_s, a_p, out);
    }
}
